// CrossAttention_412316861130
// MI455X (gfx1250) — matrix operand order repaired and checked by reading, not compiled, not run on hardware
//
#include <hip/hip_runtime.h>
#include <hip/hip_bf16.h>
#include <type_traits>

typedef __attribute__((ext_vector_type(16))) _Float16 v16h;
typedef __attribute__((ext_vector_type(8)))  _Float16 v8h;
typedef __attribute__((ext_vector_type(4)))  _Float16 v4h;
typedef __attribute__((ext_vector_type(2)))  _Float16 h2;
typedef __attribute__((ext_vector_type(8)))  float    v8f;
typedef __attribute__((ext_vector_type(4)))  float    v4f;
typedef __attribute__((ext_vector_type(4)))  int      v4i;

#define D_MODEL 1024
#define N_HEAD  16
#define HEAD_DIM 64
#define SEQ 2048
#define BATCH 2
#define MTOT (BATCH * SEQ)   // 4096 rows

// ---- CDNA5 async global->LDS staging (probe-confirmed working) --------------
#if __has_builtin(__builtin_amdgcn_global_load_async_to_lds_b128) && \
    __has_builtin(__builtin_amdgcn_s_wait_asynccnt)
#define USE_ASYNC 1
#else
#define USE_ASYNC 0
#endif

#define AS1 __attribute__((address_space(1)))
#define AS3 __attribute__((address_space(3)))

// ---------------------------------------------------------------------------
__device__ __forceinline__ v16h combine(v8h lo, v8h hi) {
  return __builtin_shufflevector(lo, hi, 0, 1, 2, 3, 4, 5, 6, 7,
                                 8, 9, 10, 11, 12, 13, 14, 15);
}

__device__ __forceinline__ h2 cvt_pk(float a, float b) {
  return __builtin_bit_cast(h2, __builtin_amdgcn_cvt_pkrtz(a, b));
}

__device__ __forceinline__ v8f wmma16(v16h a, v16h b, v8f c) {
  return __builtin_amdgcn_wmma_f32_16x16x32_f16(
      false, a, false, b, (short)0, c, false, false);
}

__device__ __forceinline__ float fast_exp2(float x) {
#if __has_builtin(__builtin_amdgcn_exp2f)
  return __builtin_amdgcn_exp2f(x);
#else
  return __builtin_exp2f(x);
#endif
}
__device__ __forceinline__ float fast_rcp(float x) {
#if __has_builtin(__builtin_amdgcn_rcpf)
  return __builtin_amdgcn_rcpf(x);
#else
  return 1.0f / x;
#endif
}

// single-instruction max (avoids clang's canonicalize self-max around fmaxf)
__device__ __forceinline__ float vmaxf(float a, float b) {
  float r;
  asm("v_max_num_f32 %0, %1, %2" : "=v"(r) : "v"(a), "v"(b));
  return r;
}

// fused DPP reduction steps: dst = op(dpp(src0), src1), one VALU inst each.
// Tree xor1, xor2, xor7, xor15 covers the 16-lane C/D row group (wave32).
__device__ __forceinline__ float red_max16(float x) {
  float t;
  asm("v_max_num_f32_dpp %0, %1, %2 quad_perm:[1,0,3,2] row_mask:0xf bank_mask:0xf bound_ctrl:1"
      : "=v"(t) : "v"(x), "v"(x));
  asm("v_max_num_f32_dpp %0, %1, %2 quad_perm:[2,3,0,1] row_mask:0xf bank_mask:0xf bound_ctrl:1"
      : "=v"(x) : "v"(t), "v"(t));
  asm("v_max_num_f32_dpp %0, %1, %2 row_half_mirror row_mask:0xf bank_mask:0xf bound_ctrl:1"
      : "=v"(t) : "v"(x), "v"(x));
  asm("v_max_num_f32_dpp %0, %1, %2 row_mirror row_mask:0xf bank_mask:0xf bound_ctrl:1"
      : "=v"(x) : "v"(t), "v"(t));
  return x;
}
__device__ __forceinline__ float red_sum16(float x) {
  float t;
  asm("v_add_f32_dpp %0, %1, %2 quad_perm:[1,0,3,2] row_mask:0xf bank_mask:0xf bound_ctrl:1"
      : "=v"(t) : "v"(x), "v"(x));
  asm("v_add_f32_dpp %0, %1, %2 quad_perm:[2,3,0,1] row_mask:0xf bank_mask:0xf bound_ctrl:1"
      : "=v"(x) : "v"(t), "v"(t));
  asm("v_add_f32_dpp %0, %1, %2 row_half_mirror row_mask:0xf bank_mask:0xf bound_ctrl:1"
      : "=v"(t) : "v"(x), "v"(x));
  asm("v_add_f32_dpp %0, %1, %2 row_mirror row_mask:0xf bank_mask:0xf bound_ctrl:1"
      : "=v"(x) : "v"(t), "v"(t));
  return x;
}

// ---------------------------------------------------------------------------
// C[M,N] = (A[M,K] * B[N,K]^T + bias) * outScale.
// Block: 256 thr (8 waves), 128x128 tile; wave tile 32x64 (2x4 frags).
// THEAD=true stores output transposed per head: [B*H][HEAD_DIM][SEQ] (for kp).
// ---------------------------------------------------------------------------
template <typename AT, typename OutT, bool THEAD>
__global__ __launch_bounds__(256)
void gemm_nt_wmma(const AT* __restrict__ A, const float* __restrict__ Bw,
                  const float* __restrict__ bias, OutT* __restrict__ C,
                  int M, int N, int K, float outScale) {
  __shared__ __align__(16) _Float16 As[128 * 48];
  __shared__ __align__(16) _Float16 BsT[32 * 136];

  constexpr bool kAsyncA = (USE_ASYNC != 0) && std::is_same<AT, _Float16>::value;

  const int t    = threadIdx.x;
  const int lane = t & 31;
  const int wave = t >> 5;                 // 0..7
  const int wm   = (wave >> 1) * 32;       // 0,32,64,96
  const int wn   = (wave & 1) * 64;        // 0,64
  const int m0   = blockIdx.y * 128;
  const int n0   = blockIdx.x * 128;
  const int g    = (lane >> 4) * 8;        // A-frag K sub-offset per lane half

  const int nb = (t >> 3) * 4;             // B staging: 4-row n block 0..124
  const int kb = (t & 7) * 4;              // B staging: 4-col k block 0..28

  v8f acc[2][4] = {};

  for (int k0 = 0; k0 < K; k0 += 32) {
    __syncthreads();
    // ---- A tile: 128 rows x 32 K, 16B granules (512 slots / 256 thr) ----
#pragma unroll
    for (int i = 0; i < 2; ++i) {
      int idx8 = t + i * 256;
      int row  = idx8 >> 2;
      int c8   = (idx8 & 3) * 8;
      const AT* src = A + (size_t)(m0 + row) * K + k0 + c8;
      _Float16* dst = &As[row * 48 + c8];
      if constexpr (kAsyncA) {
        __builtin_amdgcn_global_load_async_to_lds_b128(
            (AS1 v4i*)src, (AS3 v4i*)dst, 0, 0);
      } else if constexpr (std::is_same<AT, float>::value) {
        v4f s0 = *(const v4f*)src;
        v4f s1 = *(const v4f*)(src + 4);
        v4h lo = __builtin_shufflevector(cvt_pk(s0[0], s0[1]),
                                         cvt_pk(s0[2], s0[3]), 0, 1, 2, 3);
        v4h hi = __builtin_shufflevector(cvt_pk(s1[0], s1[1]),
                                         cvt_pk(s1[2], s1[3]), 0, 1, 2, 3);
        *(v8h*)dst = __builtin_shufflevector(lo, hi, 0, 1, 2, 3, 4, 5, 6, 7);
      } else {
        *(v8h*)dst = *(const v8h*)src;
      }
    }
    // ---- B tile: per-thread 4n x 4k block; pack n-quads -> ds_store_b64 ----
    {
      _Float16 tmp[4][4];
#pragma unroll
      for (int r = 0; r < 4; ++r) {
        const float* src = Bw + (size_t)(n0 + nb + r) * K + k0 + kb;
        v4f s = *(const v4f*)src;
        h2 lo = cvt_pk(s[0], s[1]);
        h2 hi = cvt_pk(s[2], s[3]);
        tmp[r][0] = lo[0]; tmp[r][1] = lo[1];
        tmp[r][2] = hi[0]; tmp[r][3] = hi[1];
      }
#pragma unroll
      for (int e = 0; e < 4; ++e) {
        v4h d;
        d[0] = tmp[0][e]; d[1] = tmp[1][e];
        d[2] = tmp[2][e]; d[3] = tmp[3][e];
        *(v4h*)&BsT[(kb + e) * 136 + nb] = d;
      }
    }
#if USE_ASYNC
    if constexpr (kAsyncA) __builtin_amdgcn_s_wait_asynccnt(0);
#endif
    __syncthreads();

    v16h bf[4];
#pragma unroll
    for (int j = 0; j < 4; ++j) {
      int nbase = wn + j * 16 + (lane & 15);
#pragma unroll
      for (int s = 0; s < 16; ++s)
        bf[j][s] = BsT[((s & 8) * 2 + g + (s & 7)) * 136 + nbase];
    }
#pragma unroll
    for (int i = 0; i < 2; ++i) {
      int mr = wm + i * 16 + (lane & 15);
      v16h af = combine(*(const v8h*)&As[mr * 48 + g],
                        *(const v8h*)&As[mr * 48 + 16 + g]);
#pragma unroll
      for (int j = 0; j < 4; ++j)
        acc[i][j] = wmma16(af, bf[j], acc[i][j]);
    }
  }

  const int group = lane >> 4;
  const int nn    = lane & 15;
#pragma unroll
  for (int i = 0; i < 2; ++i) {
#pragma unroll
    for (int j = 0; j < 4; ++j) {
      int ncol = n0 + wn + j * 16 + nn;
      float bv = bias[ncol];
#pragma unroll
      for (int r = 0; r < 8; ++r) {
        int mrow = m0 + wm + i * 16 + r + group * 8;
        float val = (acc[i][j][r] + bv) * outScale;
        if constexpr (THEAD) {
          // [B*H][HEAD_DIM][SEQ] layout for async K staging in attention
          int trow = mrow & (SEQ - 1);
          int bidx = mrow >> 11;           // SEQ == 2048
          int hh   = ncol >> 6;
          int feat = ncol & 63;
          C[(((size_t)bidx * N_HEAD + hh) * HEAD_DIM + feat) * SEQ + trow] =
              (OutT)val;
        } else {
          C[(size_t)mrow * N + ncol] = (OutT)val;
        }
      }
    }
  }
}

// ---------------------------------------------------------------------------
// Flash attention: block = 8 waves = 128 query rows of one (b,h).
// K arrives pre-transposed ([B*H][64][SEQ]) -> both K and V chunks are pure
// async global->LDS b128 copies, double-buffered on ASYNCcnt.
// ---------------------------------------------------------------------------
__global__ __launch_bounds__(256)
void flash_attn_wmma(const _Float16* __restrict__ Qp,
                     const _Float16* __restrict__ KpT,
                     const _Float16* __restrict__ Vp,
                     _Float16* __restrict__ Y) {
  __shared__ __align__(16) _Float16 Ks[2][64 * 48];   // [feat][key], stride 48
  __shared__ __align__(16) _Float16 Vs[2][32 * 72];   // [key][feat], stride 72
  __shared__ __align__(16) _Float16 Pl[8][16 * 48];

  const int t    = threadIdx.x;
  const int lane = t & 31;
  const int wave = t >> 5;
  const int bh   = blockIdx.y;             // b*16 + h
  const int b    = bh >> 4;
  const int q0   = blockIdx.x * 128 + wave * 16;
  const int g    = (lane >> 4) * 8;
  const size_t base =
      (size_t)b * SEQ * D_MODEL + (size_t)(bh & 15) * HEAD_DIM;

  // Q fragments (qp already carries 0.125*log2e scale): 4 global b128 loads
  v16h qf[2];
  {
    const _Float16* qrow = Qp + base + (size_t)(q0 + (lane & 15)) * D_MODEL;
#pragma unroll
    for (int kc = 0; kc < 2; ++kc)
      qf[kc] = combine(*(const v8h*)(qrow + kc * 32 + g),
                       *(const v8h*)(qrow + kc * 32 + 16 + g));
  }

  v8f o[4] = {};
  float mrow[8], lrow[8];
#pragma unroll
  for (int r = 0; r < 8; ++r) { mrow[r] = -1e30f; lrow[r] = 0.0f; }

  const int kfeat = t >> 2;                // 0..63
  const int kkey  = (t & 3) * 8;           // 0..24
  const int srow  = t >> 3;                // V: key row 0..31
  const int scol  = (t & 7) * 8;           // V: feature col 0..56

  auto stage = [&](int bufi, int j) {
    const _Float16* kg =
        KpT + ((size_t)bh * HEAD_DIM + kfeat) * SEQ + j + kkey;
    const _Float16* vg = Vp + base + (size_t)(j + srow) * D_MODEL + scol;
    _Float16* kd = &Ks[bufi][kfeat * 48 + kkey];
    _Float16* vd = &Vs[bufi][srow * 72 + scol];
#if USE_ASYNC
    __builtin_amdgcn_global_load_async_to_lds_b128((AS1 v4i*)kg, (AS3 v4i*)kd,
                                                   0, 0);
    __builtin_amdgcn_global_load_async_to_lds_b128((AS1 v4i*)vg, (AS3 v4i*)vd,
                                                   0, 0);
#else
    *(v8h*)kd = *(const v8h*)kg;
    *(v8h*)vd = *(const v8h*)vg;
#endif
  };

  stage(0, 0);
  for (int j0 = 0; j0 < SEQ; j0 += 32) {
    const int cur = (j0 >> 5) & 1;
    if (j0 + 32 < SEQ) {
      stage(cur ^ 1, j0 + 32);             // prefetch next chunk
#if USE_ASYNC
      __builtin_amdgcn_s_wait_asynccnt(2); // oldest 2 (cur chunk) complete
#endif
    } else {
#if USE_ASYNC
      __builtin_amdgcn_s_wait_asynccnt(0);
#endif
    }
    __syncthreads();

    // S(16x32) = Q K^T : two 16x16 tiles over 2 feature chunks
    v8f s0 = {}, s1 = {};
#pragma unroll
    for (int kc = 0; kc < 2; ++kc) {
      const _Float16* krow = &Ks[cur][kc * 32 * 48 + (lane & 15)];
      v16h kb0, kb1;
#pragma unroll
      for (int s = 0; s < 16; ++s) {
        int kk = ((s & 8) * 2 + g + (s & 7)) * 48;
        kb0[s] = krow[kk];
        kb1[s] = krow[kk + 16];
      }
      s0 = wmma16(qf[kc], kb0, s0);
      s1 = wmma16(qf[kc], kb1, s1);
    }

    // online softmax in base-2 (scale folded into qp); fused DPP reductions
    {
      _Float16* Pw = Pl[wave];
      const int group = lane >> 4;
      const int nn    = lane & 15;
      float alpha[8];
#pragma unroll
      for (int r = 0; r < 8; ++r) {
        float a0 = s0[r];
        float a1 = s1[r];
        float mx   = red_max16(vmaxf(a0, a1));
        float mnew = vmaxf(mrow[r], mx);
        float al   = fast_exp2(mrow[r] - mnew);
        float p0   = fast_exp2(a0 - mnew);
        float p1   = fast_exp2(a1 - mnew);
        float ps   = red_sum16(p0 + p1);
        lrow[r]  = lrow[r] * al + ps;
        mrow[r]  = mnew;
        alpha[r] = al;
        int m = r + group * 8;
        Pw[m * 48 + nn]      = (_Float16)p0;
        Pw[m * 48 + 16 + nn] = (_Float16)p1;
      }
#pragma unroll
      for (int f = 0; f < 4; ++f)
#pragma unroll
        for (int r = 0; r < 8; ++r)
          o[f][r] *= alpha[r];

      const _Float16* prow = &Pw[(lane & 15) * 48];
      v16h pf = combine(*(const v8h*)(prow + g),
                        *(const v8h*)(prow + 16 + g));

#pragma unroll
      for (int f = 0; f < 4; ++f) {
        const _Float16* vrow = &Vs[cur][f * 16 + (lane & 15)];
        v16h vf;
#pragma unroll
        for (int s = 0; s < 16; ++s)
          vf[s] = vrow[((s & 8) * 2 + g + (s & 7)) * 72];
        o[f] = wmma16(pf, vf, o[f]);
      }
    }
    __syncthreads();
  }

  const int group = lane >> 4;
  const int nn    = lane & 15;
#pragma unroll
  for (int r = 0; r < 8; ++r) {
    float inv = fast_rcp(lrow[r]);
    int m = q0 + r + group * 8;
#pragma unroll
    for (int f = 0; f < 4; ++f)
      Y[base + (size_t)m * D_MODEL + f * 16 + nn] = (_Float16)(o[f][r] * inv);
  }
}

// ---------------------------------------------------------------------------
__global__ void f32_to_f16_kernel(const float* __restrict__ x,
                                  _Float16* __restrict__ y, int n) {
  int i = (blockIdx.x * blockDim.x + threadIdx.x) * 4;
  if (i < n) {
    v4f s = *(const v4f*)(x + i);
    h2 lo = cvt_pk(s[0], s[1]);
    h2 hi = cvt_pk(s[2], s[3]);
    *(v4h*)(y + i) = __builtin_shufflevector(lo, hi, 0, 1, 2, 3);
  }
}

// ---------------------------------------------------------------------------
extern "C" void kernel_launch(void* const* d_in, const int* in_sizes, int n_in,
                              void* d_out, int out_size, void* d_ws, size_t ws_size,
                              hipStream_t stream) {
  const float* k  = (const float*)d_in[0];
  const float* q  = (const float*)d_in[1];
  const float* v  = (const float*)d_in[2];
  const float* Wk = (const float*)d_in[3];
  const float* bk = (const float*)d_in[4];
  const float* Wq = (const float*)d_in[5];
  const float* bq = (const float*)d_in[6];
  const float* Wc = (const float*)d_in[7];
  const float* bc = (const float*)d_in[8];
  float* out = (float*)d_out;

  const size_t SZ = (size_t)MTOT * D_MODEL;
  _Float16* kpT  = (_Float16*)d_ws;     // [B*H][HEAD_DIM][SEQ]
  _Float16* qp16 = kpT + SZ;            // [M][D] (pre-scaled)
  _Float16* v16  = qp16 + SZ;           // [M][D]
  _Float16* y16  = v16  + SZ;           // [M][D]

  const float qscale = 0.125f * 1.4426950408889634f;  // 1/sqrt(64) * log2(e)
  dim3 gblk(D_MODEL / 128, MTOT / 128);

  gemm_nt_wmma<float, _Float16, true><<<gblk, 256, 0, stream>>>(
      k, Wk, bk, kpT, MTOT, D_MODEL, D_MODEL, 1.0f);
  gemm_nt_wmma<float, _Float16, false><<<gblk, 256, 0, stream>>>(
      q, Wq, bq, qp16, MTOT, D_MODEL, D_MODEL, qscale);
  f32_to_f16_kernel<<<(int)(SZ / 1024), 256, 0, stream>>>(v, v16, (int)SZ);

  flash_attn_wmma<<<dim3(SEQ / 128, BATCH * N_HEAD), 256, 0, stream>>>(
      qp16, kpT, v16, y16);

  gemm_nt_wmma<_Float16, float, false><<<gblk, 256, 0, stream>>>(
      y16, Wc, bc, out, MTOT, D_MODEL, D_MODEL, 1.0f);
}
